// Attn_1185410973711
// MI455X (gfx1250) — compile-verified, matrix operand order checked by reading, not run on hardware
//
#include <hip/hip_runtime.h>
#include <math.h>

#define H   512
#define SS  2048
#define BB  32
#define K2  1024   // 2*H
#define MT  64     // s-rows per workgroup

typedef __attribute__((ext_vector_type(16))) __bf16 v16bf;
typedef __attribute__((ext_vector_type(8)))  float  v8f;
typedef __attribute__((ext_vector_type(4)))  unsigned int u32x4;

struct frag32 { u32x4 lo, hi; };   // 32 bytes == v16bf

static __device__ inline unsigned short f2bf_rne(float f) {
  unsigned u = __builtin_bit_cast(unsigned, f);
  u += 0x7FFFu + ((u >> 16) & 1u);          // round-to-nearest-even
  return (unsigned short)(u >> 16);
}

// branch-free tanh: native v_tanh_f32 on gfx1250, else exp-based
static __device__ inline float fast_tanh(float x) {
#if __has_builtin(__builtin_amdgcn_tanhf)
  return __builtin_amdgcn_tanhf(x);
#else
  float ax = fabsf(x);
  float e  = __expf(2.0f * ax);             // v_exp_f32, no branches
  float t  = 1.0f - 2.0f / (e + 1.0f);      // e->inf => t->1 (correct saturation)
  return copysignf(t, x);
#endif
}

// ---------------------------------------------------------------------------
// Repack W_e = W_attn[:, H:3H] into bf16 B-operand fragments.
// Tile t = kt*32 + nt ; per tile: 32 lanes x 16 halfs contiguous (1 KB).
// Lane half-map (mirrors 16-bit A layout): lane<16 -> K {0..7,16..23},
// lane>=16 -> K {8..15,24..31}; N = nt*16 + (lane&15).
// ---------------------------------------------------------------------------
__global__ __launch_bounds__(256) void prep_B(const float* __restrict__ W,
                                              unsigned short* __restrict__ Bws) {
  int idx  = blockIdx.x * 256 + threadIdx.x;   // 0 .. 1024*512-1
  int j    = idx & 15;
  int lane = (idx >> 4) & 31;
  int t    = idx >> 9;
  int nt   = t & 31;
  int kt   = t >> 5;
  int koff = ((lane >= 16) ? 8 : 0) + (j & 7) + ((j >= 8) ? 16 : 0);
  int n    = nt * 16 + (lane & 15);
  int k    = kt * 32 + koff;
  Bws[idx] = f2bf_rne(W[(size_t)n * (3 * H) + H + k]);
}

// bias[b,h] = output[b,:] . W_attn[h,:H] + b_attn[h]
__global__ __launch_bounds__(256) void prep_bias(const float* __restrict__ st,
                                                 const float* __restrict__ W,
                                                 const float* __restrict__ ba,
                                                 float* __restrict__ biasWs) {
  int idx = blockIdx.x * 256 + threadIdx.x;    // 0..16383
  int b = idx >> 9, h = idx & (H - 1);
  const float* o = st + (size_t)b * H;
  const float* w = W + (size_t)h * (3 * H);
  float s = 0.f;
#pragma unroll 4
  for (int k = 0; k < H; ++k) s += o[k] * w[k];
  biasWs[idx] = s + ba[h];
}

// ---------------------------------------------------------------------------
// Main: per workgroup (b, 64 s-rows). 8 waves: mSub = wave&3 (16 rows),
// nHalf = wave>>2 (256 h). Stage A in 128 KB LDS (bf16), loop n-tiles with
// full-K WMMA accumulation, fused tanh/dot-v epilogue, cross-wave combine.
// ---------------------------------------------------------------------------
__global__ __launch_bounds__(256) void attn_main(const float* __restrict__ enc,
                                                 const unsigned short* __restrict__ Bws,
                                                 const float* __restrict__ biasWs,
                                                 const float* __restrict__ vvec,
                                                 float* __restrict__ scoresWs) {
  __shared__ unsigned short As[MT * K2];        // 128 KB; reused for reduction
  const int b   = blockIdx.y;
  const int s0  = blockIdx.x * MT;
  const int tid = threadIdx.x;

  // Stage A: MT rows x 1024 f32 -> bf16 (rows are contiguous in memory)
  {
    const float4* src = (const float4*)(enc + ((size_t)b * SS + s0) * K2);
    for (int i = tid; i < (MT * K2) / 4; i += 256) {
      float4 f = src[i];
      int e = i * 4;
      As[e + 0] = f2bf_rne(f.x);
      As[e + 1] = f2bf_rne(f.y);
      As[e + 2] = f2bf_rne(f.z);
      As[e + 3] = f2bf_rne(f.w);
    }
  }
  __syncthreads();

  const int wave  = tid >> 5;
  const int lane  = tid & 31;
  const int mSub  = wave & 3;                  // 0..3 : 16-row subtile
  const int nHalf = wave >> 2;                 // 0..1 : 256-h half
  const int mrow  = mSub * 16 + (lane & 15);
  const int kbl   = (lane >= 16) ? 8 : 0;      // per-lane K base within k-tile

  float sp[8];
#pragma unroll
  for (int r = 0; r < 8; ++r) sp[r] = 0.f;

#pragma unroll 1
  for (int n = 0; n < 16; ++n) {
    const int ntg = nHalf * 16 + n;            // global n-tile 0..31
    v8f acc = {};
    const unsigned short* bt = Bws + ((size_t)ntg * 32 + lane) * 16;
#pragma unroll 4
    for (int kt = 0; kt < 32; ++kt) {
      const u32x4* ap = (const u32x4*)(As + (size_t)mrow * K2 + kt * 32 + kbl);
      frag32 af;  af.lo = ap[0];  af.hi = ap[2];     // K +0..7 and +16..23
      const u32x4* bp = (const u32x4*)(bt + (size_t)kt * (32 * 32 * 16));
      frag32 bfr; bfr.lo = bp[0]; bfr.hi = bp[1];
      v16bf av = __builtin_bit_cast(v16bf, af);
      v16bf bv = __builtin_bit_cast(v16bf, bfr);
      acc = __builtin_amdgcn_wmma_f32_16x16x32_bf16(false, av, false, bv,
                                                    (short)0, acc, false, false);
    }
    // epilogue: lane owns column h; VGPR r is row r (lanes<16) / r+8 (lanes>=16)
    int   h    = ntg * 16 + (lane & 15);
    float bias = biasWs[(size_t)b * H + h];
    float vh   = vvec[h];
#pragma unroll
    for (int r = 0; r < 8; ++r) sp[r] += fast_tanh(acc[r] + bias) * vh;
  }

  // reduce across the 16 lanes of each half (xor 1,2,4,8 stays in-half)
#pragma unroll
  for (int r = 0; r < 8; ++r) {
    float x = sp[r];
    x += __shfl_xor(x, 1);
    x += __shfl_xor(x, 2);
    x += __shfl_xor(x, 4);
    x += __shfl_xor(x, 8);
    sp[r] = x;
  }

  __syncthreads();                              // A staging no longer needed
  float* sred = (float*)As;                     // [8 waves][16 local rows]
  if ((lane & 15) == 0) {
    int half = lane >> 4;
#pragma unroll
    for (int r = 0; r < 8; ++r) sred[wave * 16 + half * 8 + r] = sp[r];
  }
  __syncthreads();
  if (tid < MT) {
    int R  = tid;                               // block-local row 0..63
    int w0 = R >> 4;                            // waves with mSub == R>>4
    float s = sred[w0 * 16 + (R & 15)] + sred[(w0 + 4) * 16 + (R & 15)];
    scoresWs[(size_t)b * SS + s0 + R] = s;
  }
}

// mask penalty + stable softmax over S per batch row
__global__ __launch_bounds__(256) void softmax_k(const float* __restrict__ scoresWs,
                                                 const int* __restrict__ mask,
                                                 float* __restrict__ out) {
  __shared__ float red[256];
  int b = blockIdx.x, tid = threadIdx.x;
  const float* sc = scoresWs + (size_t)b * SS;
  const int*   mk = mask + (size_t)b * SS;
  float vals[8];
  float mx = -1e30f;
#pragma unroll
  for (int i = 0; i < 8; ++i) {
    int s = tid + i * 256;
    float x = sc[s] - ((mk[s] == 0) ? 1000.0f : 0.0f);
    vals[i] = x;
    mx = fmaxf(mx, x);
  }
  red[tid] = mx; __syncthreads();
  for (int off = 128; off > 0; off >>= 1) {
    if (tid < off) red[tid] = fmaxf(red[tid], red[tid + off]);
    __syncthreads();
  }
  mx = red[0]; __syncthreads();
  float sum = 0.f;
#pragma unroll
  for (int i = 0; i < 8; ++i) { vals[i] = __expf(vals[i] - mx); sum += vals[i]; }
  red[tid] = sum; __syncthreads();
  for (int off = 128; off > 0; off >>= 1) {
    if (tid < off) red[tid] += red[tid + off];
    __syncthreads();
  }
  float inv = 1.0f / red[0];
#pragma unroll
  for (int i = 0; i < 8; ++i) out[(size_t)b * SS + tid + i * 256] = vals[i] * inv;
}

extern "C" void kernel_launch(void* const* d_in, const int* in_sizes, int n_in,
                              void* d_out, int out_size, void* d_ws, size_t ws_size,
                              hipStream_t stream) {
  const float* st   = (const float*)d_in[0];   // output (B,H)
  const float* enc  = (const float*)d_in[1];   // encoder_outputs (B,S,2H)
  const int*   mk   = (const int*)d_in[2];     // encoder_mask (B,S)
  const float* W    = (const float*)d_in[3];   // W_attn (H,3H)
  const float* ba   = (const float*)d_in[4];   // b_attn (H)
  const float* v    = (const float*)d_in[5];   // v (H)
  float* out = (float*)d_out;

  char* ws = (char*)d_ws;
  unsigned short* Bws     = (unsigned short*)ws;                       // 1 MB
  float*          biasWs  = (float*)(ws + (1u << 20));                 // 64 KB
  float*          scoresW = (float*)(ws + (1u << 20) + (1u << 16));    // 256 KB

  prep_B   <<<(K2 * H) / 256, 256, 0, stream>>>(W, Bws);
  prep_bias<<<(BB * H) / 256, 256, 0, stream>>>(st, W, ba, biasWs);
  attn_main<<<dim3(SS / MT, BB), 256, 0, stream>>>(enc, Bws, biasWs, v, scoresW);
  softmax_k<<<BB, 256, 0, stream>>>(scoresW, mk, out);
}
